// IntraBlock_23278722744349
// MI455X (gfx1250) — compile-verified, matrix operand order checked by reading, not run on hardware
//
#include <hip/hip_runtime.h>
#include <hip/hip_bf16.h>
#include <math.h>

// ---------------------------------------------------------------------------
// Types for CDNA5 WMMA (wave32): A/B = 16 x bf16 (8 VGPRs), C/D = 8 x f32
// ---------------------------------------------------------------------------
typedef __attribute__((ext_vector_type(16))) __bf16          bf16x16;
typedef __attribute__((ext_vector_type(8)))  float           f32x8;
typedef __attribute__((ext_vector_type(8)))  unsigned short  u16x8;

union FragU { bf16x16 v; u16x8 h[2]; };

#define DEVI __device__ __forceinline__

DEVI unsigned short f2bf(float f) {
    unsigned int u = __float_as_uint(f);
    u += 0x7FFFu + ((u >> 16) & 1u);          // round-to-nearest-even
    return (unsigned short)(u >> 16);
}

DEVI float gelu_exact(float x) {
    return 0.5f * x * (1.0f + erff(x * 0.70710678118654752f));
}

DEVI f32x8 wmma_bf16(bf16x16 a, bf16x16 b, f32x8 c) {
    // (neg_a, A, neg_b, B, c_mod, C, reuse_a, reuse_b)
    return __builtin_amdgcn_wmma_f32_16x16x32_bf16(false, a, false, b,
                                                   (short)0, c, false, false);
}

// Load one 16xK=32 bf16 A/B fragment from an LDS tile whose rows are
// `strideElem` halfwords wide (stride bytes must be a multiple of 16).
// Element e of the fragment holds k = kstep*32 + (e<8?0:16) + half*8 + (e&7),
// i.e. two contiguous 16-byte chunks.
DEVI bf16x16 ld_frag(const unsigned short* tile, int strideElem, int row,
                     int kstep, int half) {
    const char* base = (const char*)tile + (size_t)row * strideElem * 2
                     + kstep * 64 + half * 16;
    FragU f;
    f.h[0] = *(const u16x8*)(base);
    f.h[1] = *(const u16x8*)(base + 32);
    return f.v;
}

// ---------------------------------------------------------------------------
// Generic bf16 GEMM: C[M,N] = epilogue(A[M,K] @ B[K,N] + bias (+res))
// Block: 256 threads (8 waves), block tile 128x128, wave tile 32x64,
// K step 32, register-staged double-buffered LDS (1 barrier / K-step).
// EPI: 0 = f32 (+bias), 1 = bf16 (+bias), 2 = bf16 gelu(+bias),
//      3 = f32 (+bias+residual)
// ---------------------------------------------------------------------------
template<int EPI>
__global__ __launch_bounds__(256)
void gemm_bf16_kernel(const unsigned short* __restrict__ A,
                      const unsigned short* __restrict__ B,
                      const float* __restrict__ bias,
                      const float* __restrict__ res,
                      void* __restrict__ Cv,
                      int M, int N, int K)
{
    __shared__ __align__(16) unsigned short As[2][128 * 40]; // [m][k] stride 40
    __shared__ __align__(16) unsigned short Bs[2][128 * 40]; // [n][k] (B transposed)

    const int tid  = threadIdx.x;
    const int w    = tid >> 5;
    const int lane = tid & 31;
    const int half = lane >> 4;
    const int l16  = lane & 15;
    const int wm   = (w & 3) * 32;   // wave row group within block tile
    const int wn   = (w >> 2) * 64;  // wave col group within block tile
    const int mTile = blockIdx.y * 128;
    const int nTile = blockIdx.x * 128;

    f32x8 acc[2][4];
#pragma unroll
    for (int i = 0; i < 2; ++i)
#pragma unroll
        for (int j = 0; j < 4; ++j) acc[i][j] = (f32x8)0.0f;

    // staging assignments
    const int arow = tid >> 1, aseg = tid & 1;          // A: 128 rows x 2 segs of 16
    const int bk   = tid & 31, bn0  = (tid >> 5) << 4;  // B: 32 k x 8 n-chunks of 16

    const unsigned short* gA = A + (size_t)(mTile + arow) * K + aseg * 16;
    const unsigned short* gB = B + (size_t)bk * N + nTile + bn0;

    // register-staged prefetch of tile 0
    uint4 aR0, aR1;
    union BU { uint4 u[2]; unsigned short s[16]; } bR;
    {
        const uint4* ga = (const uint4*)gA;
        aR0 = ga[0]; aR1 = ga[1];
        const uint4* gb = (const uint4*)gB;
        bR.u[0] = gb[0]; bR.u[1] = gb[1];
    }

    int cur = 0;
    for (int kt = 0; kt < K; kt += 32) {
        // drain staged registers into LDS buffer `cur`
        {
            uint4* s = (uint4*)(&As[cur][arow * 40 + aseg * 16]);
            s[0] = aR0; s[1] = aR1;
#pragma unroll
            for (int i = 0; i < 16; ++i) Bs[cur][(bn0 + i) * 40 + bk] = bR.s[i];
        }
        __syncthreads();

        // issue global loads for the next K tile (in flight during compute)
        if (kt + 32 < K) {
            const uint4* ga = (const uint4*)(gA + (kt + 32));
            aR0 = ga[0]; aR1 = ga[1];
            const uint4* gb = (const uint4*)(gB + (size_t)(kt + 32) * N);
            bR.u[0] = gb[0]; bR.u[1] = gb[1];
            // pull the tile after next into GL2 (global_prefetch_b8)
            if (kt + 64 < K) {
                __builtin_prefetch(gA + (kt + 64), 0, 1);
                __builtin_prefetch(gB + (size_t)(kt + 64) * N, 0, 1);
            }
        }

        // compute: 2 A-frags x 4 B-frags = 8 WMMA per K-step
        bf16x16 aF[2];
#pragma unroll
        for (int i = 0; i < 2; ++i)
            aF[i] = ld_frag(As[cur], 40, wm + i * 16 + l16, 0, half);
#pragma unroll
        for (int j = 0; j < 4; ++j) {
            const bf16x16 bF = ld_frag(Bs[cur], 40, wn + j * 16 + l16, 0, half);
#pragma unroll
            for (int i = 0; i < 2; ++i)
                acc[i][j] = wmma_bf16(aF[i], bF, acc[i][j]);
        }
        cur ^= 1;
    }

    // Epilogue. C/D layout: VGPR r holds (row = r + 8*half, col = lane%16).
#pragma unroll
    for (int i = 0; i < 2; ++i) {
#pragma unroll
        for (int j = 0; j < 4; ++j) {
            const int col = nTile + wn + j * 16 + l16;
            const float bv = bias[col];
#pragma unroll
            for (int r = 0; r < 8; ++r) {
                const int row = mTile + wm + i * 16 + r + 8 * half;
                const size_t idx = (size_t)row * N + col;
                float v = acc[i][j][r] + bv;
                if (EPI == 0)      ((float*)Cv)[idx]          = v;
                else if (EPI == 1) ((unsigned short*)Cv)[idx] = f2bf(v);
                else if (EPI == 2) ((unsigned short*)Cv)[idx] = f2bf(gelu_exact(v));
                else               ((float*)Cv)[idx]          = v + res[idx];
            }
        }
    }
}

// ---------------------------------------------------------------------------
// Flash attention: grid (V/64, H, B), 128 threads (4 waves x 16 queries).
// Q,K,V bf16 [B*V, D]; lm = log-mask fp32 [B, V, V]; O bf16 [B*V, D]. DK = 64.
// ---------------------------------------------------------------------------
__global__ __launch_bounds__(128)
void attn_kernel(const unsigned short* __restrict__ Q,
                 const unsigned short* __restrict__ Kg,
                 const unsigned short* __restrict__ Vg,
                 const float* __restrict__ lm,
                 unsigned short* __restrict__ O,
                 int Vlen, int D)
{
    __shared__ __align__(16) unsigned short Qs[64 * 72];  // [q][d]
    __shared__ __align__(16) unsigned short Ks[64 * 72];  // [key][d]  (=B^T layout)
    __shared__ __align__(16) unsigned short Vs[64 * 72];  // [d][key]  (transposed)
    __shared__ __align__(16) unsigned short Ps[64 * 72];  // [q][key]  (wave-private rows)

    const int tid  = threadIdx.x;
    const int w    = tid >> 5;
    const int lane = tid & 31;
    const int half = lane >> 4;
    const int l16  = lane & 15;
    const int b  = blockIdx.z;
    const int h  = blockIdx.y;
    const int q0 = blockIdx.x * 64;
    const size_t tokBase = (size_t)b * Vlen;
    const int dk0 = h * 64;

    // stage Q tile 64x64
    {
        const int row = tid >> 1, seg = tid & 1;
        const uint4* g = (const uint4*)(Q + (tokBase + q0 + row) * D + dk0 + seg * 32);
        uint4* s = (uint4*)(&Qs[row * 72 + seg * 32]);
#pragma unroll
        for (int i = 0; i < 4; ++i) s[i] = g[i];
    }

    float mrow[8], lrow[8];
    f32x8 oacc[4];
#pragma unroll
    for (int r = 0; r < 8; ++r) { mrow[r] = -1e30f; lrow[r] = 0.0f; }
#pragma unroll
    for (int j = 0; j < 4; ++j) oacc[j] = (f32x8)0.0f;

    const float qscale = 0.125f;  // 1/sqrt(64)

    for (int kb = 0; kb < Vlen; kb += 64) {
        __syncthreads();  // protect Ks/Vs from previous iteration's readers
        // stage K tile [key][d]
        {
            const int row = tid >> 1, seg = tid & 1;
            const uint4* g = (const uint4*)(Kg + (tokBase + kb + row) * D + dk0 + seg * 32);
            uint4* s = (uint4*)(&Ks[row * 72 + seg * 32]);
#pragma unroll
            for (int i = 0; i < 4; ++i) s[i] = g[i];
        }
        // stage V transposed -> Vs[d][key]
        {
            const int key = tid >> 1, seg = tid & 1;
            union { uint4 u[4]; unsigned short s[32]; } tmp;
            const uint4* g = (const uint4*)(Vg + (tokBase + kb + key) * D + dk0 + seg * 32);
#pragma unroll
            for (int i = 0; i < 4; ++i) tmp.u[i] = g[i];
#pragma unroll
            for (int i = 0; i < 32; ++i) Vs[(seg * 32 + i) * 72 + key] = tmp.s[i];
        }
        __syncthreads();

        // S = Q K^T   (16 q x 64 key per wave)
        f32x8 sacc[4];
#pragma unroll
        for (int j = 0; j < 4; ++j) sacc[j] = (f32x8)0.0f;
#pragma unroll
        for (int ks = 0; ks < 2; ++ks) {
            const bf16x16 a = ld_frag(Qs, 72, w * 16 + l16, ks, half);
#pragma unroll
            for (int j = 0; j < 4; ++j) {
                const bf16x16 bb = ld_frag(Ks, 72, j * 16 + l16, ks, half);
                sacc[j] = wmma_bf16(a, bb, sacc[j]);
            }
        }

        // add log-mask, online softmax update
        float s[4][8], tmax[8];
#pragma unroll
        for (int r = 0; r < 8; ++r) tmax[r] = -1e30f;
#pragma unroll
        for (int j = 0; j < 4; ++j) {
            const int key = kb + j * 16 + l16;
#pragma unroll
            for (int r = 0; r < 8; ++r) {
                const int qrow = q0 + w * 16 + r + 8 * half;
                float v = sacc[j][r] * qscale
                        + lm[((size_t)b * Vlen + qrow) * Vlen + key];
                s[j][r] = v;
                tmax[r] = fmaxf(tmax[r], v);
            }
        }
#pragma unroll
        for (int r = 0; r < 8; ++r) {           // row max over 16 lanes
            float v = tmax[r];
#pragma unroll
            for (int m = 1; m < 16; m <<= 1) v = fmaxf(v, __shfl_xor(v, m, 32));
            tmax[r] = v;
        }
        float escale[8];
#pragma unroll
        for (int r = 0; r < 8; ++r) {
            const float mnew = fmaxf(mrow[r], tmax[r]);
            escale[r] = expf(mrow[r] - mnew);
            mrow[r] = mnew;
        }
        float rsum[8];
#pragma unroll
        for (int r = 0; r < 8; ++r) rsum[r] = 0.0f;
#pragma unroll
        for (int j = 0; j < 4; ++j)
#pragma unroll
            for (int r = 0; r < 8; ++r) {
                const float p = expf(s[j][r] - mrow[r]);
                s[j][r] = p;
                rsum[r] += p;
            }
#pragma unroll
        for (int r = 0; r < 8; ++r) {           // row sum over 16 lanes
            float v = rsum[r];
#pragma unroll
            for (int m = 1; m < 16; m <<= 1) v += __shfl_xor(v, m, 32);
            lrow[r] = lrow[r] * escale[r] + v;
        }
#pragma unroll
        for (int j = 0; j < 4; ++j)
#pragma unroll
            for (int r = 0; r < 8; ++r) oacc[j][r] *= escale[r];

        // P -> wave-private LDS rows (re-layout C -> A fragment)
#pragma unroll
        for (int j = 0; j < 4; ++j)
#pragma unroll
            for (int r = 0; r < 8; ++r)
                Ps[(w * 16 + r + 8 * half) * 72 + j * 16 + l16] = f2bf(s[j][r]);
        __builtin_amdgcn_wave_barrier();  // same-wave LDS ops stay in order

        // O += P V
#pragma unroll
        for (int ks = 0; ks < 2; ++ks) {
            const bf16x16 a = ld_frag(Ps, 72, w * 16 + l16, ks, half);
#pragma unroll
            for (int j = 0; j < 4; ++j) {
                const bf16x16 bb = ld_frag(Vs, 72, j * 16 + l16, ks, half);
                oacc[j] = wmma_bf16(a, bb, oacc[j]);
            }
        }
    }

    // finalize: O /= (l + 1e-8)
    float inv[8];
#pragma unroll
    for (int r = 0; r < 8; ++r) inv[r] = 1.0f / (lrow[r] + 1e-8f);
#pragma unroll
    for (int j = 0; j < 4; ++j) {
        const int col = dk0 + j * 16 + l16;
#pragma unroll
        for (int r = 0; r < 8; ++r) {
            const size_t tok = tokBase + q0 + w * 16 + r + 8 * half;
            O[tok * D + col] = f2bf(oacc[j][r] * inv[r]);
        }
    }
}

// ---------------------------------------------------------------------------
// Elementwise helpers
// ---------------------------------------------------------------------------
__global__ void cvt_f32_bf16_kernel(const float* __restrict__ s,
                                    unsigned short* __restrict__ d, long n)
{
    long i = (long)blockIdx.x * blockDim.x + threadIdx.x;
    if (i < n) d[i] = f2bf(s[i]);
}

__global__ void logmask_kernel(const float* __restrict__ conn,
                               float* __restrict__ lm, long n)
{
    long i = (long)blockIdx.x * blockDim.x + threadIdx.x;
    if (i < n) {
        float c = fminf(fmaxf(conn[i], 0.0f), 1.0f);
        lm[i] = logf(c + 1e-6f);
    }
}

__global__ __launch_bounds__(256)
void layernorm_kernel(const float* __restrict__ x, const float* __restrict__ g,
                      const float* __restrict__ b, float* __restrict__ y, int D)
{
    __shared__ float s1[256], s2[256];
    const int row = blockIdx.x;
    const float* xr = x + (size_t)row * D;
    float a1 = 0.0f, a2 = 0.0f;
    for (int i = threadIdx.x; i < D; i += 256) {
        const float v = xr[i];
        a1 += v; a2 += v * v;
    }
    s1[threadIdx.x] = a1; s2[threadIdx.x] = a2;
    __syncthreads();
    for (int off = 128; off > 0; off >>= 1) {
        if ((int)threadIdx.x < off) {
            s1[threadIdx.x] += s1[threadIdx.x + off];
            s2[threadIdx.x] += s2[threadIdx.x + off];
        }
        __syncthreads();
    }
    const float mean = s1[0] / (float)D;
    const float var  = s2[0] / (float)D - mean * mean;
    const float rinv = rsqrtf(var + 1e-5f);
    float* yr = y + (size_t)row * D;
    for (int i = threadIdx.x; i < D; i += 256)
        yr[i] = (xr[i] - mean) * rinv * g[i] + b[i];
}

// y_bf = bf16( yln * (1 + gb[:, :D]) + gb[:, D:] )
__global__ void film_kernel(const float* __restrict__ yln,
                            const float* __restrict__ gb,
                            unsigned short* __restrict__ ybf, int D, long n)
{
    long i = (long)blockIdx.x * blockDim.x + threadIdx.x;
    if (i < n) {
        const long row = i / D, col = i % D;
        const float g  = gb[(size_t)row * 2 * D + col];
        const float be = gb[(size_t)row * 2 * D + D + col];
        ybf[i] = f2bf(yln[i] * (1.0f + g) + be);
    }
}

// ---------------------------------------------------------------------------
// Host launcher
// ---------------------------------------------------------------------------
extern "C" void kernel_launch(void* const* d_in, const int* in_sizes, int n_in,
                              void* d_out, int out_size, void* d_ws, size_t ws_size,
                              hipStream_t stream)
{
    (void)in_sizes; (void)n_in; (void)out_size; (void)ws_size;
    const int Bb = 2, Vv = 2048, D = 1024, Hh = 16, DE = 256, DFF = 4096;
    const int NTOK = Bb * Vv;
    const int D2 = 2 * D;

    const float* x     = (const float*)d_in[0];
    const float* e     = (const float*)d_in[1];
    const float* conn  = (const float*)d_in[2];
    const float* ln1_g = (const float*)d_in[3];
    const float* ln1_b = (const float*)d_in[4];
    const float* f1w1  = (const float*)d_in[5];
    const float* f1b1  = (const float*)d_in[6];
    const float* f1w2  = (const float*)d_in[7];
    const float* f1b2  = (const float*)d_in[8];
    const float* wq    = (const float*)d_in[9];
    const float* bq    = (const float*)d_in[10];
    const float* wk    = (const float*)d_in[11];
    const float* bk    = (const float*)d_in[12];
    const float* wv    = (const float*)d_in[13];
    const float* bv    = (const float*)d_in[14];
    const float* wp    = (const float*)d_in[15];
    const float* bp    = (const float*)d_in[16];
    const float* ln2_g = (const float*)d_in[17];
    const float* ln2_b = (const float*)d_in[18];
    const float* f2w1  = (const float*)d_in[19];
    const float* f2b1  = (const float*)d_in[20];
    const float* f2w2  = (const float*)d_in[21];
    const float* f2b2  = (const float*)d_in[22];
    const float* ffw1  = (const float*)d_in[23];
    const float* ffb1  = (const float*)d_in[24];
    const float* ffw2  = (const float*)d_in[25];
    const float* ffb2  = (const float*)d_in[26];
    float* out = (float*)d_out;

    // ---- workspace layout (bump allocator, 256B aligned) ----
    char* p = (char*)d_ws;
    auto alloc = [&](size_t bytes) -> void* {
        void* r = (void*)p;
        p += (bytes + 255) & ~(size_t)255;
        return r;
    };
    typedef unsigned short bf;
    bf* f1w1b = (bf*)alloc((size_t)DE  * D2  * 2);
    bf* f1w2b = (bf*)alloc((size_t)D2  * D2  * 2);
    bf* f2w1b = (bf*)alloc((size_t)DE  * D2  * 2);
    bf* f2w2b = (bf*)alloc((size_t)D2  * D2  * 2);
    bf* wqb   = (bf*)alloc((size_t)D   * D   * 2);
    bf* wkb   = (bf*)alloc((size_t)D   * D   * 2);
    bf* wvb   = (bf*)alloc((size_t)D   * D   * 2);
    bf* wpb   = (bf*)alloc((size_t)D   * D   * 2);
    bf* ffw1b = (bf*)alloc((size_t)D   * DFF * 2);
    bf* ffw2b = (bf*)alloc((size_t)DFF * D   * 2);
    bf* e_bf  = (bf*)alloc((size_t)NTOK * DE * 2);
    float* yln = (float*)alloc((size_t)NTOK * D  * 4);   // reused for LN1/LN2
    bf*    hb  = (bf*)   alloc((size_t)NTOK * D2 * 2);   // FiLM hidden (reused)
    float* gbf = (float*)alloc((size_t)NTOK * D2 * 4);   // gamma|beta (reused)
    bf*    ybf = (bf*)   alloc((size_t)NTOK * D  * 2);   // post-FiLM acts (reused)
    bf*    Qb  = (bf*)   alloc((size_t)NTOK * D  * 2);
    bf*    Kb  = (bf*)   alloc((size_t)NTOK * D  * 2);
    bf*    Vb  = (bf*)   alloc((size_t)NTOK * D  * 2);
    bf*    Ob  = (bf*)   alloc((size_t)NTOK * D  * 2);
    float* x2  = (float*)alloc((size_t)NTOK * D  * 4);
    // lm (B*V*V*4 = 32MB) and t (NTOK*DFF*2 = 32MB) are disjoint in time: share.
    void* shared = alloc((size_t)Bb * Vv * Vv * 4);
    float* lmb = (float*)shared;
    bf*    tb  = (bf*)shared;

    auto cvt = [&](const float* s, bf* d, long n) {
        cvt_f32_bf16_kernel<<<(int)((n + 255) / 256), 256, 0, stream>>>(s, d, n);
    };
    auto gemm = [&](int EPI, const bf* A, const bf* B, const float* bias,
                    const float* res, void* C, int M, int N, int K) {
        dim3 grid(N / 128, M / 128);
        switch (EPI) {
        case 0: gemm_bf16_kernel<0><<<grid, 256, 0, stream>>>(A, B, bias, res, C, M, N, K); break;
        case 1: gemm_bf16_kernel<1><<<grid, 256, 0, stream>>>(A, B, bias, res, C, M, N, K); break;
        case 2: gemm_bf16_kernel<2><<<grid, 256, 0, stream>>>(A, B, bias, res, C, M, N, K); break;
        default: gemm_bf16_kernel<3><<<grid, 256, 0, stream>>>(A, B, bias, res, C, M, N, K); break;
        }
    };

    // ---- weight / activation conversions ----
    cvt(f1w1, f1w1b, (long)DE * D2);
    cvt(f1w2, f1w2b, (long)D2 * D2);
    cvt(f2w1, f2w1b, (long)DE * D2);
    cvt(f2w2, f2w2b, (long)D2 * D2);
    cvt(wq, wqb, (long)D * D);
    cvt(wk, wkb, (long)D * D);
    cvt(wv, wvb, (long)D * D);
    cvt(wp, wpb, (long)D * D);
    cvt(ffw1, ffw1b, (long)D * DFF);
    cvt(ffw2, ffw2b, (long)DFF * D);
    cvt(e, e_bf, (long)NTOK * DE);

    // ---- log mask ----
    {
        long n = (long)Bb * Vv * Vv;
        logmask_kernel<<<(int)((n + 255) / 256), 256, 0, stream>>>(conn, lmb, n);
    }

    // ---- LN1 + FiLM1 ----
    layernorm_kernel<<<NTOK, 256, 0, stream>>>(x, ln1_g, ln1_b, yln, D);
    gemm(2, e_bf, f1w1b, f1b1, nullptr, hb, NTOK, D2, DE);   // h1 = gelu(e@W1+b1)
    gemm(0, hb, f1w2b, f1b2, nullptr, gbf, NTOK, D2, D2);    // gb1
    {
        long n = (long)NTOK * D;
        film_kernel<<<(int)((n + 255) / 256), 256, 0, stream>>>(yln, gbf, ybf, D, n);
    }

    // ---- QKV projections (bf16 out) ----
    gemm(1, ybf, wqb, bq, nullptr, Qb, NTOK, D, D);
    gemm(1, ybf, wkb, bk, nullptr, Kb, NTOK, D, D);
    gemm(1, ybf, wvb, bv, nullptr, Vb, NTOK, D, D);

    // ---- attention ----
    {
        dim3 ag(Vv / 64, Hh, Bb);
        attn_kernel<<<ag, 128, 0, stream>>>(Qb, Kb, Vb, lmb, Ob, Vv, D);
    }

    // ---- output projection + residual: x2 = x + O@wp + bp ----
    gemm(3, Ob, wpb, bp, x, x2, NTOK, D, D);

    // ---- LN2 + FiLM2 ----
    layernorm_kernel<<<NTOK, 256, 0, stream>>>(x2, ln2_g, ln2_b, yln, D);
    gemm(2, e_bf, f2w1b, f2b1, nullptr, hb, NTOK, D2, DE);
    gemm(0, hb, f2w2b, f2b2, nullptr, gbf, NTOK, D2, D2);
    {
        long n = (long)NTOK * D;
        film_kernel<<<(int)((n + 255) / 256), 256, 0, stream>>>(yln, gbf, ybf, D, n);
    }

    // ---- FFN: out = x2 + gelu(y2@W1+b1)@W2 + b2 ----
    gemm(2, ybf, ffw1b, ffb1, nullptr, tb, NTOK, DFF, D);    // t (bf16, reuses lm space)
    gemm(3, tb, ffw2b, ffb2, x2, out, NTOK, D, DFF);
}
